// Pre_Callback_Post_Projector_28862180229897
// MI455X (gfx1250) — compile-verified, matrix operand order checked by reading, not run on hardware
//
#include <hip/hip_runtime.h>
#include <cstdint>

// ---------------------------------------------------------------------------
// ViT-Det windowed-attention block for MI455X (gfx1250, wave32, WMMA).
// All GEMMs run on v_wmma_f32_16x16x32_f16 (f16 in, f32 accumulate) with
// 2x2 register blocking per wave (4 WMMAs per K-chunk, fragment reuse).
// GEMM A-tiles are staged global->LDS with CDNA5 async-DMA
// (global_load_async_to_lds_b128 + s_wait_asynccnt).
// Attention is fused per (window, head) with the whole 208x224 score matrix
// resident in LDS (CDNA5: 320KB/WGP), so attn scores never touch HBM.
// ---------------------------------------------------------------------------

typedef __attribute__((ext_vector_type(16))) _Float16 v16h;
typedef __attribute__((ext_vector_type(8)))  _Float16 v8h;
typedef __attribute__((ext_vector_type(4)))  _Float16 v4h;
typedef __attribute__((ext_vector_type(8)))  float    v8f;

#define BATCH   8
#define HH      64
#define WW2     64
#define DIMC    768
#define NHEAD   12
#define HD      64
#define WS2     14
#define NTOK    196      // WS*WS tokens per window
#define NPAD    208      // 13 * 16  (M padding for WMMA tiles)
#define KPAD    224      // 7  * 32  (K padding for P@V)
#define MLPC    3072
#define NWIN    200      // BATCH * 5 * 5 windows
#define TOKENS  32768    // BATCH * H * W

__device__ __forceinline__ v16h cat8(v8h lo, v8h hi) {
  return __builtin_shufflevector(lo, hi, 0,1,2,3,4,5,6,7,8,9,10,11,12,13,14,15);
}

// LDS byte offset of a __shared__ pointer: aperture mapping puts the
// addrspace(3) offset in the low 32 bits of the flat address.
__device__ __forceinline__ uint32_t lds_off(const void* p) {
  return (uint32_t)(uintptr_t)p;
}

// ---------------------------------------------------------------------------
// f32 -> f16 weight conversion
// ---------------------------------------------------------------------------
__global__ void cvt_kernel(const float* __restrict__ src,
                           _Float16* __restrict__ dst, int n) {
  int i = blockIdx.x * 256 + threadIdx.x;
  if (i < n) dst[i] = (_Float16)src[i];
}

// ---------------------------------------------------------------------------
// LayerNorm over DIM=768, one wave32 per row, output f16
// ---------------------------------------------------------------------------
__global__ __launch_bounds__(256)
void ln_kernel(const float* __restrict__ x, const float* __restrict__ w,
               const float* __restrict__ b, _Float16* __restrict__ out,
               int rows) {
  int lane = threadIdx.x & 31;
  int wave = threadIdx.x >> 5;
  int row  = blockIdx.x * 8 + wave;
  if (row >= rows) return;
  const float* p = x + (size_t)row * DIMC;
  float s = 0.f, s2 = 0.f;
  for (int c = lane; c < DIMC; c += 32) { float v = p[c]; s += v; s2 += v * v; }
  for (int off = 16; off; off >>= 1) {
    s  += __shfl_xor(s,  off, 32);
    s2 += __shfl_xor(s2, off, 32);
  }
  float mean = s * (1.f / DIMC);
  float var  = s2 * (1.f / DIMC) - mean * mean;
  float rinv = rsqrtf(var + 1e-5f);
  for (int c = lane; c < DIMC; c += 32)
    out[(size_t)row * DIMC + c] = (_Float16)((p[c] - mean) * rinv * w[c] + b[c]);
}

// ---------------------------------------------------------------------------
// Generic WMMA GEMM: C[M,N] = A[M,K] (f16) * W[K,N] (f16) + bias, + epilogue
//   block: 256 threads = 8 waves as 2(M) x 4(N); each wave owns a 32x32
//   output tile (2x2 WMMA tiles) => block tile 64(M) x 128(N).
//   A tile (64x32) staged with async global->LDS DMA (EXEC-masked lanes;
//   invalid/padded rows stay at the pre-zeroed value).
//   B tile staged transposed [n][k] (stride 40 to spread banks).
//   mode 0: QKV   A = LN1 output gathered through window partition (zero pad)
//                 out f16 [win*196+n][2304]
//   mode 1: PROJ  A = attention out [win*196+n][768]
//                 out f32 d_out[b,th,tw,:] = x + acc + bias  (unpartition)
//   mode 2: MLP1  A = LN2 output [tok][768]; GELU; out f16 [tok][3072]
//   mode 3: MLP2  A = hidden [tok][3072];    d_out += acc + bias
// ---------------------------------------------------------------------------
__global__ __launch_bounds__(256)
void gemm_kernel(int mode,
                 const _Float16* __restrict__ A,
                 const _Float16* __restrict__ Wt,    // [K][N] f16
                 const float*    __restrict__ bias,  // [N]
                 _Float16* __restrict__ outH,
                 float*    __restrict__ outF,
                 const float* __restrict__ resid,
                 int K, int N) {
  __shared__ _Float16 ldsA[64 * 32];    // [m][k]
  __shared__ _Float16 ldsB[128 * 40];   // transposed [n][k], stride 40

  const int tid   = threadIdx.x;
  const int lane  = tid & 31;
  const int wave  = tid >> 5;
  const int wm    = wave >> 2;          // 0..1  (M position of wave)
  const int wn    = wave & 3;           // 0..3  (N position of wave)
  const int Nbase = blockIdx.x * 128;
  const int Mbase = blockIdx.y * 64;
  const int win   = blockIdx.z;

  // A staging: thread t copies 16B: row = t/4, 8 f16 at (t%4)*8
  const int arow = tid >> 2;            // 0..63
  const int aseg = (tid & 3) * 8;       // 0,8,16,24
  const _Float16* aptr = nullptr;       // row base (null => zero-padded row)
  {
    int n = Mbase + arow;
    if (mode == 0 || mode == 1) {
      if (n < NTOK) {
        if (mode == 1) {
          aptr = A + (size_t)(win * NTOK + n) * DIMC;
        } else {
          int b  = win / 25, rem = win % 25;
          int wh = rem / 5,  wc  = rem % 5;
          int th = wh * WS2 + n / WS2;
          int tw = wc * WS2 + n % WS2;
          if (th < HH && tw < WW2)
            aptr = A + (size_t)((b * HH + th) * WW2 + tw) * DIMC;
        }
      }
    } else {
      aptr = A + (size_t)(Mbase + arow) * K;
    }
  }
  const uint32_t a_lds = lds_off(&ldsA[arow * 32 + aseg]);

  // B staging: thread t handles n = t/2, 16 strided K elements at (t%2)*16
  const int bn = tid >> 1;              // 0..127
  const int bk = (tid & 1) * 16;        // 0 or 16

  // pre-zero A tile so rows skipped by the EXEC-masked async DMA stay zero
  {
    v4h z = {};
    *(v4h*)&ldsA[tid * 8]     = z;
    *(v4h*)&ldsA[tid * 8 + 4] = z;
  }
  __syncthreads();

  v8f acc[2][2] = {};
  for (int K0 = 0; K0 < K; K0 += 32) {
    __syncthreads();
    // ---- A tile: CDNA5 async global->LDS DMA (16B per active lane) ----
    if (aptr) {
      uint64_t ga = (uint64_t)(uintptr_t)(aptr + K0 + aseg);
      asm volatile("global_load_async_to_lds_b128 %0, %1, off"
                   :: "v"(a_lds), "v"(ga) : "memory");
    }
    // ---- B tile: transposed scalar staging ----
#pragma unroll
    for (int j = 0; j < 16; ++j)
      ldsB[bn * 40 + bk + j] = Wt[(size_t)(K0 + bk + j) * N + Nbase + bn];
    asm volatile("s_wait_asynccnt 0" ::: "memory");
    __syncthreads();

    // ---- fragments + 2x2 WMMA ----
    const int fr = lane & 15;
    const int kb = (lane >> 4) * 8;
    v16h af[2], bf[2];
#pragma unroll
    for (int ti = 0; ti < 2; ++ti) {
      const _Float16* pa = &ldsA[(wm * 32 + ti * 16 + fr) * 32 + kb];
      af[ti] = cat8(*(const v8h*)pa, *(const v8h*)(pa + 16));
    }
#pragma unroll
    for (int tj = 0; tj < 2; ++tj) {
      const _Float16* pb = &ldsB[(wn * 32 + tj * 16 + fr) * 40 + kb];
      bf[tj] = cat8(*(const v8h*)pb, *(const v8h*)(pb + 16));
    }
#pragma unroll
    for (int ti = 0; ti < 2; ++ti)
#pragma unroll
      for (int tj = 0; tj < 2; ++tj)
        acc[ti][tj] = __builtin_amdgcn_wmma_f32_16x16x32_f16(
            false, af[ti], false, bf[tj], (short)0, acc[ti][tj], false, false);
  }

  // ---- epilogue ----
#pragma unroll
  for (int tj = 0; tj < 2; ++tj) {
    const int col = Nbase + wn * 32 + tj * 16 + (lane & 15);
    const float bv = bias[col];
#pragma unroll
    for (int ti = 0; ti < 2; ++ti) {
#pragma unroll
      for (int r = 0; r < 8; ++r) {
        int mrow = Mbase + wm * 32 + ti * 16 + r + (lane >> 4) * 8;
        float v = acc[ti][tj][r] + bv;
        if (mode == 0) {
          if (mrow < NTOK)
            outH[(size_t)(win * NTOK + mrow) * (3 * DIMC) + col] = (_Float16)v;
        } else if (mode == 1) {
          if (mrow < NTOK) {
            int b  = win / 25, rem = win % 25;
            int wh = rem / 5,  wc  = rem % 5;
            int th = wh * WS2 + mrow / WS2;
            int tw = wc * WS2 + mrow % WS2;
            if (th < HH && tw < WW2) {
              size_t idx = (size_t)((b * HH + th) * WW2 + tw) * DIMC + col;
              outF[idx] = resid[idx] + v;
            }
          }
        } else if (mode == 2) {
          float g = 0.5f * v * (1.0f + erff(v * 0.70710678118f)); // exact GELU
          outH[(size_t)mrow * MLPC + col] = (_Float16)g;
        } else {
          size_t idx = (size_t)mrow * DIMC + col;
          outF[idx] = outF[idx] + v;                               // residual
        }
      }
    }
  }
}

// ---------------------------------------------------------------------------
// Fused attention per (window, head): 2400 blocks x 256 threads.
// LDS: q[208x64] k[208x64] vT[64x224] S[208x224] (f16), rel tables (f32).
// S = softmax(scale*q@k^T + rel_h + rel_w); O = S @ v.
// ---------------------------------------------------------------------------
__global__ __launch_bounds__(256)
void attn_kernel(const _Float16* __restrict__ qkv,    // [win*196][2304]
                 const float* __restrict__ rph,       // [27][64]
                 const float* __restrict__ rpw,       // [27][64]
                 _Float16* __restrict__ outp) {       // [win*196][768]
  extern __shared__ __align__(16) char smem[];
  _Float16* q_s = (_Float16*)smem;                    // NPAD*HD
  _Float16* k_s = q_s + NPAD * HD;                    // NPAD*HD
  _Float16* vT  = k_s + NPAD * HD;                    // HD*KPAD
  _Float16* S   = vT + HD * KPAD;                     // NPAD*KPAD
  float* relh = (float*)(S + NPAD * KPAD);            // NPAD*WS2
  float* relw = relh + NPAD * WS2;                    // NPAD*WS2

  const int tid  = threadIdx.x;
  const int lane = tid & 31;
  const int wave = tid >> 5;
  const int head = blockIdx.x;
  const int win  = blockIdx.y;
  const size_t base = (size_t)win * NTOK * (3 * DIMC) + head * HD;

  // stage q, k (zero-pad rows >= 196)
  for (int i = tid; i < NPAD * HD; i += 256) {
    int n = i >> 6, c = i & 63;
    _Float16 qv = (_Float16)0.f, kv = (_Float16)0.f;
    if (n < NTOK) {
      qv = qkv[base + (size_t)n * (3 * DIMC) + c];
      kv = qkv[base + (size_t)n * (3 * DIMC) + DIMC + c];
    }
    q_s[i] = qv; k_s[i] = kv;
  }
  // stage v transposed [c][n] (zero-pad cols >= 196)
  for (int i = tid; i < HD * KPAD; i += 256) {
    int c = i / KPAD, n = i % KPAD;
    _Float16 vv = (_Float16)0.f;
    if (n < NTOK) vv = qkv[base + (size_t)n * (3 * DIMC) + 2 * DIMC + c];
    vT[i] = vv;
  }
  // zero S pad columns 208..223 (K padding for P@V)
  for (int i = tid; i < NPAD * 16; i += 256)
    S[(i >> 4) * KPAD + 208 + (i & 15)] = (_Float16)0.f;
  __syncthreads();

  // decomposed rel-pos tables: relh[i][t] = sum_c q[i][c]*Rh[hi-t+13][c]
  for (int e = tid; e < NPAD * WS2; e += 256) {
    int i = e / WS2, t = e % WS2;
    float sh = 0.f, sw = 0.f;
    if (i < NTOK) {
      int hi = i / WS2, wi = i % WS2;
      const float* ph = rph + (size_t)(hi - t + WS2 - 1) * HD;
      const float* pw = rpw + (size_t)(wi - t + WS2 - 1) * HD;
      for (int c = 0; c < HD; ++c) {
        float qv = (float)q_s[i * HD + c];
        sh += qv * ph[c];
        sw += qv * pw[c];
      }
    }
    relh[e] = sh; relw[e] = sw;
  }
  __syncthreads();

  const float scale = 0.125f;  // HD^-0.5

  // S tiles: 13x13 WMMA tiles, wave-strided
  for (int t = wave; t < 13 * 13; t += 8) {
    int mt = t / 13, nt = t % 13;
    v8f acc = {};
#pragma unroll
    for (int kk = 0; kk < 2; ++kk) {
      int fr = lane & 15;
      int kb = kk * 32 + (lane >> 4) * 8;
      v8h alo = *(const v8h*)&q_s[(mt * 16 + fr) * HD + kb];
      v8h ahi = *(const v8h*)&q_s[(mt * 16 + fr) * HD + kb + 16];
      v8h blo = *(const v8h*)&k_s[(nt * 16 + fr) * HD + kb];
      v8h bhi = *(const v8h*)&k_s[(nt * 16 + fr) * HD + kb + 16];
      acc = __builtin_amdgcn_wmma_f32_16x16x32_f16(false, cat8(alo, ahi),
                                                   false, cat8(blo, bhi),
                                                   (short)0, acc, false, false);
    }
#pragma unroll
    for (int r = 0; r < 8; ++r) {
      int i = mt * 16 + r + (lane >> 4) * 8;
      int j = nt * 16 + (lane & 15);
      float v = 0.f;
      if (j < NTOK)
        v = acc[r] * scale + relh[i * WS2 + j / WS2] + relw[i * WS2 + j % WS2];
      S[i * KPAD + j] = (_Float16)v;   // pad cols (>=196) get 0
    }
  }
  __syncthreads();

  // row softmax (one wave32 per row)
  for (int row = wave; row < NTOK; row += 8) {
    float m = -1e30f;
    for (int j = lane; j < NTOK; j += 32)
      m = fmaxf(m, (float)S[row * KPAD + j]);
    for (int off = 16; off; off >>= 1) m = fmaxf(m, __shfl_xor(m, off, 32));
    float s = 0.f;
    for (int j = lane; j < NTOK; j += 32)
      s += __expf((float)S[row * KPAD + j] - m);
    for (int off = 16; off; off >>= 1) s += __shfl_xor(s, off, 32);
    float inv = 1.f / s;
    for (int j = lane; j < NTOK; j += 32)
      S[row * KPAD + j] = (_Float16)(__expf((float)S[row * KPAD + j] - m) * inv);
  }
  __syncthreads();

  // O = P @ V : 13 x 4 WMMA tiles, K padded to 224
  for (int t = wave; t < 13 * 4; t += 8) {
    int mt = t >> 2, nt = t & 3;
    v8f acc = {};
#pragma unroll
    for (int kk = 0; kk < 7; ++kk) {
      int fr = lane & 15;
      int kb = kk * 32 + (lane >> 4) * 8;
      v8h alo = *(const v8h*)&S[(mt * 16 + fr) * KPAD + kb];
      v8h ahi = *(const v8h*)&S[(mt * 16 + fr) * KPAD + kb + 16];
      v8h blo = *(const v8h*)&vT[(nt * 16 + fr) * KPAD + kb];
      v8h bhi = *(const v8h*)&vT[(nt * 16 + fr) * KPAD + kb + 16];
      acc = __builtin_amdgcn_wmma_f32_16x16x32_f16(false, cat8(alo, ahi),
                                                   false, cat8(blo, bhi),
                                                   (short)0, acc, false, false);
    }
#pragma unroll
    for (int r = 0; r < 8; ++r) {
      int i = mt * 16 + r + (lane >> 4) * 8;
      int d = nt * 16 + (lane & 15);
      if (i < NTOK)
        outp[(size_t)(win * NTOK + i) * DIMC + head * HD + d] = (_Float16)acc[r];
    }
  }
}

// ---------------------------------------------------------------------------
// Host launcher
// ---------------------------------------------------------------------------
extern "C" void kernel_launch(void* const* d_in, const int* in_sizes, int n_in,
                              void* d_out, int out_size, void* d_ws,
                              size_t ws_size, hipStream_t stream) {
  (void)in_sizes; (void)n_in; (void)out_size; (void)ws_size;
  const float* x      = (const float*)d_in[0];
  const float* n1_w   = (const float*)d_in[1];
  const float* n1_b   = (const float*)d_in[2];
  const float* qkv_w  = (const float*)d_in[3];
  const float* qkv_b  = (const float*)d_in[4];
  const float* proj_w = (const float*)d_in[5];
  const float* proj_b = (const float*)d_in[6];
  const float* rp_h   = (const float*)d_in[7];
  const float* rp_w   = (const float*)d_in[8];
  const float* n2_w   = (const float*)d_in[9];
  const float* n2_b   = (const float*)d_in[10];
  const float* lin1_w = (const float*)d_in[11];
  const float* lin1_b = (const float*)d_in[12];
  const float* lin2_w = (const float*)d_in[13];
  const float* lin2_b = (const float*)d_in[14];
  float* out = (float*)d_out;

  // workspace layout (bytes)
  char* ws = (char*)d_ws;
  constexpr size_t OFF_XN   = 0;                         // LN out f16 (50.3MB)
  constexpr size_t OFF_QKV  = 50331648ULL;               // qkv / mlp hidden f16
  constexpr size_t OFF_ATT  = OFF_QKV + 201326592ULL;    // attn out f16 (60.2MB)
  constexpr size_t OFF_WQKV = OFF_ATT + 60211200ULL;
  constexpr size_t OFF_WPRJ = OFF_WQKV + (size_t)DIMC * 3 * DIMC * 2;
  constexpr size_t OFF_WL1  = OFF_WPRJ + (size_t)DIMC * DIMC * 2;
  constexpr size_t OFF_WL2  = OFF_WL1 + (size_t)DIMC * MLPC * 2;

  _Float16* bufXN  = (_Float16*)(ws + OFF_XN);
  _Float16* bufQKV = (_Float16*)(ws + OFF_QKV);
  _Float16* bufATT = (_Float16*)(ws + OFF_ATT);
  _Float16* wQKV   = (_Float16*)(ws + OFF_WQKV);
  _Float16* wPRJ   = (_Float16*)(ws + OFF_WPRJ);
  _Float16* wL1    = (_Float16*)(ws + OFF_WL1);
  _Float16* wL2    = (_Float16*)(ws + OFF_WL2);

  // weight conversion f32 -> f16
  cvt_kernel<<<(DIMC * 3 * DIMC + 255) / 256, 256, 0, stream>>>(qkv_w, wQKV, DIMC * 3 * DIMC);
  cvt_kernel<<<(DIMC * DIMC + 255) / 256, 256, 0, stream>>>(proj_w, wPRJ, DIMC * DIMC);
  cvt_kernel<<<(DIMC * MLPC + 255) / 256, 256, 0, stream>>>(lin1_w, wL1, DIMC * MLPC);
  cvt_kernel<<<(MLPC * DIMC + 255) / 256, 256, 0, stream>>>(lin2_w, wL2, MLPC * DIMC);

  // LN1
  ln_kernel<<<TOKENS / 8, 256, 0, stream>>>(x, n1_w, n1_b, bufXN, TOKENS);

  // QKV GEMM: [200 win x 196 (padded to 256)][768] x [768][2304]; block tile 64x128
  gemm_kernel<<<dim3(18, 4, NWIN), 256, 0, stream>>>(
      0, bufXN, wQKV, qkv_b, bufQKV, nullptr, nullptr, DIMC, 3 * DIMC);

  // fused attention: one block per (head, window); ~198KB dynamic LDS
  constexpr size_t SMEM =
      (size_t)(2 * NPAD * HD + HD * KPAD + NPAD * KPAD) * sizeof(_Float16) +
      (size_t)(2 * NPAD * WS2) * sizeof(float);   // = 198400 bytes
  attn_kernel<<<dim3(NHEAD, NWIN), 256, SMEM, stream>>>(bufQKV, rp_h, rp_w, bufATT);

  // proj GEMM + window-unpartition + residual -> d_out
  gemm_kernel<<<dim3(6, 4, NWIN), 256, 0, stream>>>(
      1, bufATT, wPRJ, proj_b, nullptr, out, x, DIMC, DIMC);

  // LN2 over (x + attn)
  ln_kernel<<<TOKENS / 8, 256, 0, stream>>>(out, n2_w, n2_b, bufXN, TOKENS);

  // MLP1 + GELU
  gemm_kernel<<<dim3(24, TOKENS / 64, 1), 256, 0, stream>>>(
      2, bufXN, wL1, lin1_b, bufQKV, nullptr, nullptr, DIMC, MLPC);

  // MLP2 + residual accumulate into d_out
  gemm_kernel<<<dim3(6, TOKENS / 64, 1), 256, 0, stream>>>(
      3, bufQKV, wL2, lin2_b, nullptr, out, nullptr, MLPC, DIMC);
}
